// TransformerBlock_65085934403864
// MI455X (gfx1250) — matrix operand order repaired and checked by reading, not compiled, not run on hardware
//
#include <hip/hip_runtime.h>
#include <math.h>

#define TT 4096
#define CC 1024
#define HH 16
#define DD 64

typedef __attribute__((ext_vector_type(16))) __bf16 v16bf;
typedef __attribute__((ext_vector_type(8)))  float  v8f;

union FragBF { v16bf v; unsigned u[8]; uint4 q[2]; };

__device__ __forceinline__ unsigned short f2bf(float f) {
  unsigned u = __float_as_uint(f);
  unsigned r = u + 0x7fffu + ((u >> 16) & 1u);
  return (unsigned short)(r >> 16);
}

// Async global->LDS copy of 16 bytes per lane (GVS mode: saddr + 32-bit voffset).
__device__ __forceinline__ void async_copy16(unsigned lds_addr, unsigned gofs,
                                             const unsigned short* base) {
  asm volatile("global_load_async_to_lds_b128 %0, %1, %2"
               :: "v"(lds_addr), "v"(gofs), "s"(base) : "memory");
}

// ---------------------------------------------------------------------------
// fp32 weights -> bf16 in WMMA-B-fragment-major layout.
// Fragment (kt, ntile) = B[kt*32 .. +31][ntile*16 .. +15] stored as
// 32 lanes x 16 contiguous bf16: lane = h*16 + (n&15) with h=(k>>4)&1,
// j = k&15 within the lane's 32-byte chunk.  One wave's fragment load is then
// two coalesced global b128 per lane (1KB contiguous per fragment).
// ---------------------------------------------------------------------------
__global__ void k_swizzle_w(const float* __restrict__ in,
                            unsigned short* __restrict__ out, int K, int N) {
  int idx = blockIdx.x * 256 + threadIdx.x;
  int total = K * N;
  int stride = gridDim.x * 256;
  for (; idx < total; idx += stride) {
    int k = idx / N, n = idx % N;
    int kt = k >> 5, h = (k >> 4) & 1, j = k & 15;
    int ntile = n >> 4, nl = n & 15;
    size_t o = (((size_t)kt * (N >> 4) + ntile) * 32 + (h * 16 + nl)) * 16 + j;
    out[o] = f2bf(in[idx]);
  }
}

// ---------------------------------------------------------------------------
// LayerNorm over C=1024, one row per block, bf16 output
// ---------------------------------------------------------------------------
__global__ __launch_bounds__(256) void k_layernorm_bf16(
    const float* __restrict__ x, const float* __restrict__ g,
    const float* __restrict__ b, unsigned short* __restrict__ out) {
  __shared__ float red1[256];
  __shared__ float red2[256];
  int row = blockIdx.x;
  const float* xr = x + (size_t)row * CC;
  float vals[4];
  float s1 = 0.f, s2 = 0.f;
  #pragma unroll
  for (int j = 0; j < 4; j++) {
    float v = xr[threadIdx.x + j * 256];
    vals[j] = v; s1 += v; s2 += v * v;
  }
  red1[threadIdx.x] = s1; red2[threadIdx.x] = s2;
  __syncthreads();
  for (int off = 128; off > 0; off >>= 1) {
    if (threadIdx.x < off) {
      red1[threadIdx.x] += red1[threadIdx.x + off];
      red2[threadIdx.x] += red2[threadIdx.x + off];
    }
    __syncthreads();
  }
  float mean = red1[0] * (1.0f / CC);
  float var  = red2[0] * (1.0f / CC) - mean * mean;
  float rs = rsqrtf(var + 1e-5f);
  #pragma unroll
  for (int j = 0; j < 4; j++) {
    int c = threadIdx.x + j * 256;
    out[(size_t)row * CC + c] = f2bf((vals[j] - mean) * rs * g[c] + b[c]);
  }
}

// ---------------------------------------------------------------------------
// bf16 WMMA GEMM.  A: row-major activations, async-LDS double buffered.
// B: pre-swizzled fragment-major weights, streamed L2->registers (no LDS),
// software-pipelined one k-step ahead in named register buffers (no dynamic
// indexing -> no scratch spills).  k-loop manually unrolled by 2.
// out = op(A[MxK] @ B[KxN] + bias [+ resid])
// mode 0: bf16 out;  mode 1: bf16 out w/ exact GELU;  mode 2: fp32 out + resid
// Block tile 64x128, k-step 32; 8 waves (2x4), each wave 32x32.
// nk = K/32 is even for all shapes used here (K = 1024 or 4096).
// ---------------------------------------------------------------------------
__global__ __launch_bounds__(256) void k_gemm_bf16(
    const unsigned short* __restrict__ A, const unsigned short* __restrict__ Bsw,
    const float* __restrict__ bias, const float* __restrict__ resid,
    unsigned short* __restrict__ outb, float* __restrict__ outf,
    int M, int N, int K, int mode) {
  const int LDA = 40;   // elements; 80B rows -> 16B-aligned async dests
  __shared__ __align__(16) unsigned short As[2][64 * 40];   // [m][k]

  int tid = threadIdx.x;
  int wave = tid >> 5, lane = tid & 31;
  int ln = lane & 15, lh = lane >> 4;
  int m0 = blockIdx.x * 64, n0 = blockIdx.y * 128;
  int wm = (wave >> 2) * 32, wn = (wave & 3) * 32;
  int ntiles = N >> 4;

  v8f zero = {0, 0, 0, 0, 0, 0, 0, 0};
  v8f acc[2][2];
  #pragma unroll
  for (int i = 0; i < 2; i++)
    #pragma unroll
    for (int j = 0; j < 2; j++) acc[i][j] = zero;

  int arow = tid >> 2;          // 0..63
  int acol = (tid & 3) * 8;     // 0..24

  auto issueA = [&](int k0, int buf) {
    unsigned ldsA = (unsigned)(unsigned long long)&As[buf][arow * LDA + acol];
    unsigned aofs = (unsigned)((m0 + arow) * K + k0 + acol) * 2u;
    async_copy16(ldsA, aofs, A);
  };
  int ntile0 = (n0 + wn) >> 4;
  auto loadB = [&](int kt, FragBF* bf) {
    const unsigned short* p =
        Bsw + (((size_t)kt * ntiles + ntile0) * 32 + lane) * 16;
    bf[0].q[0] = *(const uint4*)p;
    bf[0].q[1] = *(const uint4*)(p + 8);
    bf[1].q[0] = *(const uint4*)(p + 512);      // next 16-wide n-tile
    bf[1].q[1] = *(const uint4*)(p + 520);
  };
  auto compute = [&](const unsigned short* asb, FragBF (&bf)[2]) {
    FragBF af[2];
    #pragma unroll
    for (int i = 0; i < 2; i++) {
      const unsigned short* ap = asb + (wm + i * 16 + ln) * LDA;
      #pragma unroll
      for (int v = 0; v < 8; v++) {
        int ko = (lh << 4) + (v << 1);
        af[i].u[v] = *(const unsigned*)&ap[ko];
      }
    }
    #pragma unroll
    for (int i = 0; i < 2; i++)
      #pragma unroll
      for (int j = 0; j < 2; j++)
        acc[i][j] = __builtin_amdgcn_wmma_f32_16x16x32_bf16(
            false, af[i].v, false, bf[j].v, (short)0, acc[i][j], false, false);
  };

  FragBF b0[2], b1[2];
  int nk = K / 32;              // even
  issueA(0, 0);
  loadB(0, b0);
  for (int kt = 0; kt < nk; kt += 2) {
    // ---- step kt: compute on As[0]/b0, stage kt+1 into As[1]/b1 ----
    __syncthreads();                        // As[1] free for overwrite
    issueA((kt + 1) * 32, 1);
    loadB(kt + 1, b1);
    if (kt + 2 < nk)
      __builtin_prefetch(A + (size_t)(m0 + arow) * K + (kt + 2) * 32 + acol, 0, 3);
    asm volatile("s_wait_asynccnt 0x1" ::: "memory");   // tile kt landed
    __syncthreads();
    compute(&As[0][0], b0);

    // ---- step kt+1: compute on As[1]/b1, stage kt+2 into As[0]/b0 ----
    __syncthreads();                        // As[0] free for overwrite
    if (kt + 2 < nk) {
      issueA((kt + 2) * 32, 0);
      loadB(kt + 2, b0);
      asm volatile("s_wait_asynccnt 0x1" ::: "memory"); // tile kt+1 landed
    } else {
      asm volatile("s_wait_asynccnt 0x0" ::: "memory");
    }
    __syncthreads();
    compute(&As[1][0], b1);
  }

  #pragma unroll
  for (int i = 0; i < 2; i++) {
    int rbase = m0 + wm + i * 16 + lh * 8;
    #pragma unroll
    for (int j = 0; j < 2; j++) {
      int col = n0 + wn + j * 16 + ln;
      float bv = bias[col];
      #pragma unroll
      for (int r = 0; r < 8; r++) {
        int row = rbase + r;
        float v = acc[i][j][r] + bv;
        if (mode == 2) {
          outf[(size_t)row * N + col] = v + resid[(size_t)row * N + col];
        } else {
          if (mode == 1) v = 0.5f * v * (1.0f + erff(v * 0.70710678118f));
          outb[(size_t)row * N + col] = f2bf(v);
        }
      }
    }
  }
}

// ---------------------------------------------------------------------------
// Causal flash attention. Grid (T/64, H), 128 threads (4 waves).
// Wave w handles 16 queries; WG shares 32-key K/V LDS stage (async) per iter.
// qkv: bf16 [T][3C]; attn out: bf16 [T][C].
// ---------------------------------------------------------------------------
__global__ __launch_bounds__(128) void k_attention(
    const unsigned short* __restrict__ qkv, unsigned short* __restrict__ attn) {
  const int LDK = 72, LDV = 72, LDP = 34;
  __shared__ __align__(16) unsigned short Ks[32 * 72];     // [key][d]
  __shared__ __align__(16) unsigned short Vs[32 * 72];     // [key][d]
  __shared__ __align__(16) unsigned short Pw[4][16 * 34];  // per-wave P staging

  int tid = threadIdx.x;
  int wave = tid >> 5, lane = tid & 31;
  int ln = lane & 15, lh = lane >> 4;
  int head = blockIdx.y;
  int q0blk = blockIdx.x * 64;
  int q0 = q0blk + wave * 16;
  const int C3 = 3 * CC;

  // Preload Q fragments (16x64 = two k-steps), A-fragment layout
  FragBF qf[2];
  {
    const unsigned short* qp = qkv + (size_t)(q0 + ln) * C3 + head * DD;
    #pragma unroll
    for (int kd = 0; kd < 2; kd++)
      #pragma unroll
      for (int v = 0; v < 8; v++) {
        int ko = kd * 32 + (lh << 4) + (v << 1);
        qf[kd].u[v] = *(const unsigned*)&qp[ko];
      }
  }

  v8f zero = {0, 0, 0, 0, 0, 0, 0, 0};
  v8f accO[4];
  float mrow[8], lrow[8];
  #pragma unroll
  for (int nt = 0; nt < 4; nt++) accO[nt] = zero;
  #pragma unroll
  for (int r = 0; r < 8; r++) { mrow[r] = -3.0e38f; lrow[r] = 0.0f; }

  int nkb = (q0blk + 64) / 32;
  int lkey = tid >> 2;          // 0..31
  int ld0  = (tid & 3) * 16;    // 0..48

  for (int kb = 0; kb < nkb; kb++) {
    __syncthreads();            // previous tile fully consumed
    {  // async K/V stage: 4x b128 per thread
      int key = kb * 32 + lkey;
      unsigned kofs = (unsigned)((size_t)key * C3 + CC + head * DD + ld0) * 2u;
      unsigned vofs = kofs + CC * 2u;
      unsigned ldsK = (unsigned)(unsigned long long)&Ks[lkey * LDK + ld0];
      unsigned ldsV = (unsigned)(unsigned long long)&Vs[lkey * LDV + ld0];
      async_copy16(ldsK,       kofs,       qkv);
      async_copy16(ldsK + 16u, kofs + 16u, qkv);
      async_copy16(ldsV,       vofs,       qkv);
      async_copy16(ldsV + 16u, vofs + 16u, qkv);
    }
    asm volatile("s_wait_asynccnt 0x0" ::: "memory");
    __syncthreads();
    if (kb * 32 > q0 + 15) continue;  // wave-uniform: below the diagonal band

    // S = (Q @ K^T) for two 16-key subtiles
    v8f s[2];
    #pragma unroll
    for (int t = 0; t < 2; t++) {
      s[t] = zero;
      #pragma unroll
      for (int kd = 0; kd < 2; kd++) {
        FragBF kf;
        const unsigned short* kpr = &Ks[(t * 16 + ln) * LDK + kd * 32];
        #pragma unroll
        for (int v = 0; v < 8; v++) {
          int ko = (lh << 4) + (v << 1);
          kf.u[v] = *(const unsigned*)&kpr[ko];
        }
        s[t] = __builtin_amdgcn_wmma_f32_16x16x32_bf16(
            false, qf[kd].v, false, kf.v, (short)0, s[t], false, false);
      }
    }

    // scale + causal mask + online softmax stats
    float mnew[8];
    #pragma unroll
    for (int r = 0; r < 8; r++) {
      int qrow = q0 + lh * 8 + r;
      float best = -3.0e38f;
      #pragma unroll
      for (int t = 0; t < 2; t++) {
        int key = kb * 32 + t * 16 + ln;
        float v = s[t][r] * 0.125f;
        if (key > qrow) v = -3.0e38f;
        s[t][r] = v;
        best = fmaxf(best, v);
      }
      #pragma unroll
      for (int off = 1; off < 16; off <<= 1)
        best = fmaxf(best, __shfl_xor(best, off, 32));
      mnew[r] = fmaxf(mrow[r], best);
    }
    #pragma unroll
    for (int r = 0; r < 8; r++) {
      float alpha = __expf(mrow[r] - mnew[r]);
      float psum = 0.0f;
      #pragma unroll
      for (int t = 0; t < 2; t++) {
        float p = __expf(s[t][r] - mnew[r]);
        s[t][r] = p;
        psum += p;
      }
      #pragma unroll
      for (int off = 1; off < 16; off <<= 1)
        psum += __shfl_xor(psum, off, 32);
      lrow[r] = lrow[r] * alpha + psum;
      mrow[r] = mnew[r];
      #pragma unroll
      for (int nt = 0; nt < 4; nt++) accO[nt][r] *= alpha;
    }

    // stage P (D-layout) into LDS as [16 rows][32 keys] bf16
    #pragma unroll
    for (int r = 0; r < 8; r++) {
      int m = lh * 8 + r;
      Pw[wave][m * LDP + ln]      = f2bf(s[0][r]);
      Pw[wave][m * LDP + 16 + ln] = f2bf(s[1][r]);
    }
    asm volatile("s_wait_dscnt 0x0" ::: "memory");

    // regather P as A fragment, multiply by V
    FragBF pf;
    {
      const unsigned short* pp = &Pw[wave][ln * LDP];
      #pragma unroll
      for (int v = 0; v < 8; v++) {
        int ko = (lh << 4) + (v << 1);
        pf.u[v] = *(const unsigned*)&pp[ko];
      }
    }
    #pragma unroll
    for (int nt = 0; nt < 4; nt++) {
      int col = nt * 16 + ln;
      FragBF vf;
      const unsigned short* vp = &Vs[col];
      #pragma unroll
      for (int v = 0; v < 8; v++) {
        int kb0 = (lh << 4) + (v << 1);
        unsigned lo = vp[kb0 * LDV];
        unsigned hi = vp[(kb0 + 1) * LDV];
        vf.u[v] = lo | (hi << 16);
      }
      accO[nt] = __builtin_amdgcn_wmma_f32_16x16x32_bf16(
          false, pf.v, false, vf.v, (short)0, accO[nt], false, false);
    }
  }

  #pragma unroll
  for (int nt = 0; nt < 4; nt++)
    #pragma unroll
    for (int r = 0; r < 8; r++) {
      int row = q0 + lh * 8 + r;
      int col = head * DD + nt * 16 + ln;
      attn[(size_t)row * CC + col] = f2bf(accO[nt][r] / lrow[r]);
    }
}

// ---------------------------------------------------------------------------
extern "C" void kernel_launch(void* const* d_in, const int* in_sizes, int n_in,
                              void* d_out, int out_size, void* d_ws, size_t ws_size,
                              hipStream_t stream) {
  (void)in_sizes; (void)n_in; (void)out_size; (void)ws_size;
  const float* x    = (const float*)d_in[0];
  const float* ln1g = (const float*)d_in[1];
  const float* ln1b = (const float*)d_in[2];
  const float* Wqkv = (const float*)d_in[3];
  const float* bqkv = (const float*)d_in[4];
  const float* Wo   = (const float*)d_in[5];
  const float* bo   = (const float*)d_in[6];
  const float* ln2g = (const float*)d_in[7];
  const float* ln2b = (const float*)d_in[8];
  const float* W1   = (const float*)d_in[9];
  const float* b1   = (const float*)d_in[10];
  const float* W2   = (const float*)d_in[11];
  const float* b2   = (const float*)d_in[12];
  float* out = (float*)d_out;

  char* ws = (char*)d_ws;
  size_t off = 0;
  auto carve = [&](size_t bytes) -> char* {
    char* p = ws + off;
    off += (bytes + 255) & ~(size_t)255;
    return p;
  };
  unsigned short* Wqkvb = (unsigned short*)carve((size_t)CC * 3 * CC * 2);
  unsigned short* Wob   = (unsigned short*)carve((size_t)CC * CC * 2);
  unsigned short* W1b   = (unsigned short*)carve((size_t)CC * 4 * CC * 2);
  unsigned short* W2b   = (unsigned short*)carve((size_t)4 * CC * CC * 2);
  unsigned short* xnb   = (unsigned short*)carve((size_t)TT * CC * 2);
  unsigned short* qkvb  = (unsigned short*)carve((size_t)TT * 3 * CC * 2);
  unsigned short* attnb = (unsigned short*)carve((size_t)TT * CC * 2);
  float*          x2    = (float*)         carve((size_t)TT * CC * 4);
  unsigned short* hb    = (unsigned short*)carve((size_t)TT * CC * 2);
  unsigned short* actb  = (unsigned short*)carve((size_t)TT * 4 * CC * 2);

  // weight convert + swizzle into WMMA-fragment-major layout
  k_swizzle_w<<<2048, 256, 0, stream>>>(Wqkv, Wqkvb, CC, 3 * CC);
  k_swizzle_w<<<2048, 256, 0, stream>>>(Wo,   Wob,   CC, CC);
  k_swizzle_w<<<2048, 256, 0, stream>>>(W1,   W1b,   CC, 4 * CC);
  k_swizzle_w<<<2048, 256, 0, stream>>>(W2,   W2b,   4 * CC, CC);

  // LN1
  k_layernorm_bf16<<<TT, 256, 0, stream>>>(x, ln1g, ln1b, xnb);
  // QKV projection: [4096,1024] @ [1024,3072]
  k_gemm_bf16<<<dim3(TT / 64, (3 * CC) / 128), 256, 0, stream>>>(
      xnb, Wqkvb, bqkv, nullptr, qkvb, nullptr, TT, 3 * CC, CC, 0);
  // attention
  k_attention<<<dim3(TT / 64, HH), 128, 0, stream>>>(qkvb, attnb);
  // output projection + residual 1
  k_gemm_bf16<<<dim3(TT / 64, CC / 128), 256, 0, stream>>>(
      attnb, Wob, bo, x, nullptr, x2, TT, CC, CC, 2);
  // LN2
  k_layernorm_bf16<<<TT, 256, 0, stream>>>(x2, ln2g, ln2b, hb);
  // MLP up + exact GELU
  k_gemm_bf16<<<dim3(TT / 64, (4 * CC) / 128), 256, 0, stream>>>(
      hb, W1b, b1, nullptr, actb, nullptr, TT, 4 * CC, CC, 1);
  // MLP down + residual 2 -> d_out
  k_gemm_bf16<<<dim3(TT / 64, CC / 128), 256, 0, stream>>>(
      actb, W2b, b2, x2, nullptr, out, TT, CC, 4 * CC, 2);
}
